// Net_85598698209651
// MI455X (gfx1250) — compile-verified, matrix operand order checked by reading, not run on hardware
//
#include <hip/hip_runtime.h>
#include <cstdint>

// ---------------------------------------------------------------------------
// Types / constants
// ---------------------------------------------------------------------------
typedef __attribute__((ext_vector_type(16))) __bf16 bf16x16;
typedef __attribute__((ext_vector_type(8)))  float  f32x8;

#define NNODES 5000
#define NPAD   5120           // multiple of 512; K/M/N tile friendly
#define EDIM   256
#define DOUT   256
#define NSCALE 4

// ---------------------------------------------------------------------------
// bf16 helpers (round-to-nearest-even)
// ---------------------------------------------------------------------------
__device__ __forceinline__ uint16_t f2bf(float f) {
    union { float f; uint32_t u; } v; v.f = f;
    uint32_t r = v.u + 0x7FFFu + ((v.u >> 16) & 1u);
    return (uint16_t)(r >> 16);
}
__device__ __forceinline__ float bf2f(uint16_t b) {
    union { uint32_t u; float f; } v; v.u = ((uint32_t)b) << 16;
    return v.f;
}

// ---------------------------------------------------------------------------
// Wave-level 16x16 WMMA bf16 tile GEMM:  D(16x16,f32) = A(16xK) * B(Kx16)
// A row-major (lda in elems); B stored transposed (row n = K contiguous elems
// of column n). Lane L holds 16 consecutive K values of row/col (L&15),
// K-half selected by (L>>4) — contiguous k order, used identically for A and B.
// ---------------------------------------------------------------------------
__device__ __forceinline__ f32x8 wmma_bf16_tile(const uint16_t* __restrict__ A, int lda, int arow,
                                                const uint16_t* __restrict__ B, int ldb, int bcol,
                                                int ktot)
{
    const int lane = threadIdx.x & 31;
    const int ln = lane & 15, hi = lane >> 4;
    const uint16_t* ap = A + (size_t)(arow + ln) * (size_t)lda + hi * 16;
    const uint16_t* bp = B + (size_t)(bcol + ln) * (size_t)ldb + hi * 16;
    f32x8 acc = {};
    for (int kk = 0; kk < ktot; kk += 32) {
        __builtin_prefetch(ap + kk + 256, 0, 0);   // global_prefetch_b8
        __builtin_prefetch(bp + kk + 256, 0, 0);
        bf16x16 a = *(const bf16x16*)(const void*)(ap + kk);
        bf16x16 b = *(const bf16x16*)(const void*)(bp + kk);
        acc = __builtin_amdgcn_wmma_f32_16x16x32_bf16(
                  false, a, false, b, (short)0, acc, false, false);
    }
    return acc;
}

// ---------------------------------------------------------------------------
// Small utility kernels
// ---------------------------------------------------------------------------
__global__ void k_zero_f32(float* p, int n) {
    for (int i = blockIdx.x * 256 + threadIdx.x; i < n; i += gridDim.x * 256) p[i] = 0.f;
}
__global__ void k_zero_u16(uint16_t* p, size_t n) {
    for (size_t i = (size_t)blockIdx.x * 256 + threadIdx.x; i < n; i += (size_t)gridDim.x * 256) p[i] = 0;
}

// Vb[n][e] = bf16(V[n][e]);  Vkb[k][n][e] = bf16(V[n][e] * clk[e][SCALE_K[k]])
__global__ void k_prep_scales(const float* __restrict__ V, const float* __restrict__ clk,
                              uint16_t* __restrict__ Vkb, uint16_t* __restrict__ Vb)
{
    const int sk[NSCALE] = {31, 23, 15, 7};
    int total = NPAD * EDIM;
    for (int idx = blockIdx.x * 256 + threadIdx.x; idx < total; idx += gridDim.x * 256) {
        int n = idx / EDIM, e = idx % EDIM;
        float v = (n < NNODES) ? V[(size_t)n * EDIM + e] : 0.f;
        Vb[(size_t)n * EDIM + e] = f2bf(v);
        for (int k = 0; k < NSCALE; ++k)
            Vkb[((size_t)k * NPAD + n) * EDIM + e] = f2bf(v * clk[e * 32 + sk[k]]);
    }
}

// Weight transpose+bf16:  W[k][c][o] -> WT[o][k*Cin + c]
__global__ void k_wt(const float* __restrict__ W, uint16_t* __restrict__ WT,
                     int Kmats, int Cin, int Cout)
{
    int ktot = Kmats * Cin;
    int total = ktot * Cout;
    for (int idx = blockIdx.x * 256 + threadIdx.x; idx < total; idx += gridDim.x * 256) {
        int o = idx % Cout;
        int t = idx / Cout;            // t = k*Cin + c
        WT[(size_t)o * ktot + t] = f2bf(W[idx]);
    }
}

// ---------------------------------------------------------------------------
// Win build: tile of T = Vk * V^T ; M = (t*a[j])^2 ; store q = M^2 as bf16.
// colsq[j] += sum_i M^2.  (The column normalization (M/||col||)^2 = q/colsq
// is folded into the h^T operand of the propagation GEMM later — no second
// pass over the 210MB Win stack needed.)
// ---------------------------------------------------------------------------
__global__ void k_gemm_win(const uint16_t* __restrict__ Vk, const uint16_t* __restrict__ Vb,
                           const float* __restrict__ Avec,
                           uint16_t* __restrict__ Win, float* __restrict__ colsq)
{
    int wave = threadIdx.x >> 5;
    int jbase = blockIdx.x * 16;
    int ibase = (blockIdx.y * 8 + wave) * 16;
    f32x8 acc = wmma_bf16_tile(Vk, EDIM, ibase, Vb, EDIM, jbase, EDIM);
    int lane = threadIdx.x & 31, ln = lane & 15, hi = lane >> 4;
    int j = jbase + ln;
    float aj = (j < NNODES) ? Avec[j] : 0.f;
    float part = 0.f;
    #pragma unroll
    for (int r = 0; r < 8; ++r) {
        int i = ibase + hi * 8 + r;
        float m = acc[r] * aj; m = m * m;      // M = (t*a_j)^2
        float q = m * m;                       // q = M^2
        Win[(size_t)i * NPAD + j] = f2bf(q);
        part += q;                             // sum of M^2 -> col norm^2
    }
    if (j < NNODES) atomicAdd(&colsq[j], part);
}

// ---------------------------------------------------------------------------
// fmaxmin: per-column min/max over N rows
// ---------------------------------------------------------------------------
__global__ void k_colminmax(const float* __restrict__ Y, int ld,
                            float* __restrict__ mn, float* __restrict__ mx)
{
    int c = blockIdx.x;
    __shared__ float smn[256], smx[256];
    float vmn = 3.4e38f, vmx = -3.4e38f;
    for (int n = threadIdx.x; n < NNODES; n += 256) {
        float v = Y[(size_t)n * ld + c];
        vmn = fminf(vmn, v); vmx = fmaxf(vmx, v);
    }
    smn[threadIdx.x] = vmn; smx[threadIdx.x] = vmx;
    __syncthreads();
    for (int s = 128; s > 0; s >>= 1) {
        if (threadIdx.x < s) {
            smn[threadIdx.x] = fminf(smn[threadIdx.x], smn[threadIdx.x + s]);
            smx[threadIdx.x] = fmaxf(smx[threadIdx.x], smx[threadIdx.x + s]);
        }
        __syncthreads();
    }
    if (threadIdx.x == 0) { mn[c] = smn[0]; mx[c] = smx[0]; }
}

// Apply fmaxmin; emit bf16 h (row-major, Hcat block 0) and FOUR per-scale
// h^T operands pre-divided by max(sqrt(colsq_k[n]), eps)^2 (folded Win norm).
__global__ void k_apply_fmm(const float* __restrict__ Y, int ld, int C,
                            const float* __restrict__ mn, const float* __restrict__ mx,
                            const float* __restrict__ colsq,
                            uint16_t* __restrict__ hrow, int ldh,
                            uint16_t* __restrict__ hTs)
{
    int total = NPAD * C;
    for (int idx = blockIdx.x * 256 + threadIdx.x; idx < total; idx += gridDim.x * 256) {
        int n = idx / C, c = idx % C;
        float v = 0.f;
        if (n < NNODES) {
            float lo = mn[c], hi = mx[c];
            v = (Y[(size_t)n * ld + c] - lo) / (hi - lo);
        }
        hrow[(size_t)n * ldh + c] = f2bf(v);
        #pragma unroll
        for (int k = 0; k < NSCALE; ++k) {
            float col = sqrtf(colsq[k * NPAD + n]);
            float d = fmaxf(col, 1e-12f);
            hTs[((size_t)k * C + c) * NPAD + n] = f2bf(v / (d * d));
        }
    }
}

// ---------------------------------------------------------------------------
// Propagation: one wave computes a full 16 x C strip of Win_k @ h'.
// A (Win, streamed from HBM) is loaded ONCE per K-step and reused across
// CT = C/16 independent accumulators; B (h'^T, ~1MB) stays L2-resident.
// ---------------------------------------------------------------------------
template<int CT>
__global__ void k_gemm_prop(const uint16_t* __restrict__ Win, const uint16_t* __restrict__ hTs,
                            int C, uint16_t* __restrict__ Hcat, int ldh)
{
    int k = blockIdx.y;                       // scale index
    int wave = threadIdx.x >> 5;
    int ibase = (blockIdx.x * 8 + wave) * 16;
    int lane = threadIdx.x & 31, ln = lane & 15, hi = lane >> 4;

    const uint16_t* ap = Win + (size_t)k * NPAD * NPAD
                             + (size_t)(ibase + ln) * NPAD + hi * 16;
    const uint16_t* bp = hTs + (size_t)k * C * NPAD
                             + (size_t)ln * NPAD + hi * 16;

    f32x8 acc[CT];
    #pragma unroll
    for (int t = 0; t < CT; ++t) acc[t] = (f32x8){};

    for (int kk = 0; kk < NPAD; kk += 32) {
        __builtin_prefetch(ap + kk + 512, 0, 0);   // stream-ahead on Win
        bf16x16 a = *(const bf16x16*)(const void*)(ap + kk);
        #pragma unroll
        for (int t = 0; t < CT; ++t) {
            bf16x16 b = *(const bf16x16*)(const void*)(bp + (size_t)t * 16 * NPAD + kk);
            acc[t] = __builtin_amdgcn_wmma_f32_16x16x32_bf16(
                         false, a, false, b, (short)0, acc[t], false, false);
        }
    }

    #pragma unroll
    for (int t = 0; t < CT; ++t) {
        size_t colOff = (size_t)(k + 1) * C + t * 16 + ln;
        #pragma unroll
        for (int r = 0; r < 8; ++r) {
            int i = ibase + hi * 8 + r;
            Hcat[(size_t)i * ldh + colOff] = f2bf(acc[t][r]);
        }
    }
}

// ---------------------------------------------------------------------------
// Channel-mix GEMM: Y = [elu](Aacts @ Wstack + bias), f32 out
// ---------------------------------------------------------------------------
__global__ void k_gemm_mix(const uint16_t* __restrict__ Aacts, int lda,
                           const uint16_t* __restrict__ WT, int ktot,
                           const float* __restrict__ bias,
                           float* __restrict__ Y, int ldy, int do_elu)
{
    int wave = threadIdx.x >> 5;
    int obase = blockIdx.x * 16;
    int ibase = (blockIdx.y * 8 + wave) * 16;
    f32x8 acc = wmma_bf16_tile(Aacts, lda, ibase, WT, ktot, obase, ktot);
    int lane = threadIdx.x & 31, ln = lane & 15, hi = lane >> 4;
    int o = obase + ln;
    float b = bias ? bias[o] : 0.f;
    #pragma unroll
    for (int r = 0; r < 8; ++r) {
        int i = ibase + hi * 8 + r;
        float v = acc[r] + b;
        if (do_elu) v = (v > 0.f) ? v : (expf(v) - 1.f);
        Y[(size_t)i * ldy + o] = v;
    }
}

// des -> (d_out tail, bf16 operand)
__global__ void k_des_out(const float* __restrict__ Y, float* __restrict__ des_out,
                          uint16_t* __restrict__ desb)
{
    int total = NPAD * DOUT;
    for (int idx = blockIdx.x * 256 + threadIdx.x; idx < total; idx += gridDim.x * 256) {
        int n = idx / DOUT, c = idx % DOUT;
        float v = (n < NNODES) ? Y[(size_t)n * DOUT + c] : 0.f;
        if (n < NNODES) des_out[(size_t)n * DOUT + c] = v;
        desb[(size_t)n * DOUT + c] = f2bf(v);
    }
}

// logits = des @ fc2_w + b2  (guarded stores into [N, N] f32)
__global__ void k_gemm_logits(const uint16_t* __restrict__ desb, const uint16_t* __restrict__ W2T,
                              const float* __restrict__ b2, float* __restrict__ out)
{
    int wave = threadIdx.x >> 5;
    int jbase = blockIdx.x * 16;
    int ibase = (blockIdx.y * 8 + wave) * 16;
    f32x8 acc = wmma_bf16_tile(desb, DOUT, ibase, W2T, DOUT, jbase, DOUT);
    int lane = threadIdx.x & 31, ln = lane & 15, hi = lane >> 4;
    int j = jbase + ln;
    if (j >= NNODES) return;
    float b = b2[j];
    #pragma unroll
    for (int r = 0; r < 8; ++r) {
        int i = ibase + hi * 8 + r;
        if (i < NNODES) out[(size_t)i * NNODES + j] = acc[r] + b;
    }
}

// in-place row log_softmax over [N, N]
__global__ void k_logsoftmax(float* __restrict__ out)
{
    float* p = out + (size_t)blockIdx.x * NNODES;
    __shared__ float red[256];
    float m = -3.4e38f;
    for (int j = threadIdx.x; j < NNODES; j += 256) m = fmaxf(m, p[j]);
    red[threadIdx.x] = m; __syncthreads();
    for (int s = 128; s > 0; s >>= 1) {
        if (threadIdx.x < s) red[threadIdx.x] = fmaxf(red[threadIdx.x], red[threadIdx.x + s]);
        __syncthreads();
    }
    m = red[0]; __syncthreads();
    float sum = 0.f;
    for (int j = threadIdx.x; j < NNODES; j += 256) sum += expf(p[j] - m);
    red[threadIdx.x] = sum; __syncthreads();
    for (int s = 128; s > 0; s >>= 1) {
        if (threadIdx.x < s) red[threadIdx.x] += red[threadIdx.x + s];
        __syncthreads();
    }
    float lse = m + logf(red[0]);
    for (int j = threadIdx.x; j < NNODES; j += 256) p[j] -= lse;
}

// ---------------------------------------------------------------------------
// Host driver
// ---------------------------------------------------------------------------
extern "C" void kernel_launch(void* const* d_in, const int* in_sizes, int n_in,
                              void* d_out, int out_size, void* d_ws, size_t ws_size,
                              hipStream_t stream)
{
    (void)in_sizes; (void)n_in; (void)out_size; (void)ws_size;
    const float* x     = (const float*)d_in[0];   // [N,128]
    const float* V     = (const float*)d_in[1];   // [N,256]
    const float* Avec  = (const float*)d_in[2];   // [N,1]
    const float* clk   = (const float*)d_in[3];   // [256,32]
    const float* Wc[6] = {(const float*)d_in[4], (const float*)d_in[5], (const float*)d_in[6],
                          (const float*)d_in[7], (const float*)d_in[8], (const float*)d_in[9]};
    const float* fc1w  = (const float*)d_in[10];  // [128,256]
    const float* fc1b  = (const float*)d_in[11];  // [256]
    const float* fc2w  = (const float*)d_in[12];  // [256,5000]
    const float* fc2b  = (const float*)d_in[13];  // [5000]

    // ---- workspace carve (256B aligned segments) ----
    char* wp = (char*)d_ws;
    auto carve = [&](size_t bytes) -> void* {
        void* r = (void*)wp; wp += (bytes + 255) & ~(size_t)255; return r;
    };
    uint16_t* WinBuf = (uint16_t*)carve((size_t)NSCALE * NPAD * NPAD * 2);  // ~210 MB bf16 (M^2)
    float*    colsq  = (float*)   carve((size_t)NSCALE * NPAD * 4);
    uint16_t* Vkb    = (uint16_t*)carve((size_t)NSCALE * NPAD * EDIM * 2);
    uint16_t* Vb     = (uint16_t*)carve((size_t)NPAD * EDIM * 2);
    uint16_t* Hcat   = (uint16_t*)carve((size_t)NPAD * 640 * 2);
    uint16_t* hTs    = (uint16_t*)carve((size_t)NSCALE * 128 * NPAD * 2);   // 4 scaled h^T
    float*    Ybuf   = (float*)   carve((size_t)NPAD * DOUT * 4);
    uint16_t* desb   = (uint16_t*)carve((size_t)NPAD * DOUT * 2);
    float*    colmn  = (float*)   carve(256 * 4);
    float*    colmx  = (float*)   carve(256 * 4);
    uint16_t* WT[6];
    for (int l = 0; l < 6; ++l) WT[l] = (uint16_t*)carve((size_t)5 * 128 * 128 * 2);
    uint16_t* fc1T   = (uint16_t*)carve((size_t)DOUT * 128 * 2);
    uint16_t* fc2T   = (uint16_t*)carve((size_t)5008 * DOUT * 2);

    const dim3 TB(256);
    const int  TI8 = NPAD / (16 * 8);   // 40 row-tile groups (8 waves each)

    // ---- Win build (stores q = M^2; column norms in colsq) ----
    k_zero_f32<<<dim3(64), TB, 0, stream>>>(colsq, NSCALE * NPAD);
    k_prep_scales<<<dim3(NPAD * EDIM / 256), TB, 0, stream>>>(V, clk, Vkb, Vb);
    for (int k = 0; k < NSCALE; ++k)
        k_gemm_win<<<dim3(NPAD / 16, TI8), TB, 0, stream>>>(
            Vkb + (size_t)k * NPAD * EDIM, Vb, Avec,
            WinBuf + (size_t)k * NPAD * NPAD, colsq + k * NPAD);

    // ---- weights to bf16, transposed ----
    const int Cin[6]  = {128, 96, 96, 96, 96, 96};
    const int Cout[6] = {96, 96, 96, 96, 96, 128};
    for (int l = 0; l < 6; ++l)
        k_wt<<<dim3(256), TB, 0, stream>>>(Wc[l], WT[l], 5, Cin[l], Cout[l]);
    k_wt<<<dim3(128), TB, 0, stream>>>(fc1w, fc1T, 1, 128, DOUT);
    k_zero_u16<<<dim3(1024), TB, 0, stream>>>(fc2T, (size_t)5008 * DOUT);
    k_wt<<<dim3(4096), TB, 0, stream>>>(fc2w, fc2T, 1, DOUT, NNODES);

    // ---- input fmaxmin -> h0 (+ 4 norm-folded h^T operands) ----
    k_colminmax<<<dim3(128), TB, 0, stream>>>(x, 128, colmn, colmx);
    k_apply_fmm<<<dim3(2560), TB, 0, stream>>>(x, 128, 128, colmn, colmx, colsq,
                                               Hcat, 5 * 128, hTs);

    // ---- 6 mgconv layers ----
    for (int l = 0; l < 6; ++l) {
        int ci = Cin[l], co = Cout[l];
        int ldh = 5 * ci;
        if (ci == 128)
            k_gemm_prop<8><<<dim3(TI8, NSCALE), TB, 0, stream>>>(WinBuf, hTs, ci, Hcat, ldh);
        else
            k_gemm_prop<6><<<dim3(TI8, NSCALE), TB, 0, stream>>>(WinBuf, hTs, ci, Hcat, ldh);
        k_gemm_mix<<<dim3(co / 16, TI8), TB, 0, stream>>>(Hcat, ldh, WT[l], 5 * ci,
                                                          nullptr, Ybuf, co, 1);
        k_colminmax<<<dim3(co), TB, 0, stream>>>(Ybuf, co, colmn, colmx);
        k_apply_fmm<<<dim3((NPAD * co + 255) / 256), TB, 0, stream>>>(
            Ybuf, co, co, colmn, colmx, colsq, Hcat, 5 * co, hTs);
    }

    // ---- fc1 + elu -> des ----
    k_gemm_mix<<<dim3(DOUT / 16, TI8), TB, 0, stream>>>(Hcat, 5 * 128, fc1T, 128,
                                                        fc1b, Ybuf, DOUT, 1);
    float* out_ls  = (float*)d_out;
    float* out_des = out_ls + (size_t)NNODES * NNODES;
    k_des_out<<<dim3(NPAD * DOUT / 256), TB, 0, stream>>>(Ybuf, out_des, desb);

    // ---- logits + log_softmax ----
    k_gemm_logits<<<dim3(5008 / 16, TI8), TB, 0, stream>>>(desb, fc2T, fc2b, out_ls);
    k_logsoftmax<<<dim3(NNODES), TB, 0, stream>>>(out_ls);
}
